// Attention_87711822118915
// MI455X (gfx1250) — matrix operand order repaired and checked by reading, not compiled, not run on hardware
//
#include <hip/hip_runtime.h>
#include <stdint.h>

typedef _Float16 half_t;
typedef __attribute__((ext_vector_type(16))) _Float16 v16h;
typedef __attribute__((ext_vector_type(8)))  _Float16 v8h;
typedef __attribute__((ext_vector_type(2)))  __fp16   f16x2;  // cvt_pkrtz result type
typedef __attribute__((ext_vector_type(8)))  float    v8f;
typedef __attribute__((ext_vector_type(4)))  float    v4f;

#define TQ 4096
#define TV 4096
#define DH 64
#define LOG2E 1.44269504088896340736f
#define BIG2  1.44269504e9f  /* 1e9 * log2(e): penalty in log2 domain */

__device__ __forceinline__ v16h concat8(v8h lo, v8h hi) {
  v16h r;
#pragma unroll
  for (int i = 0; i < 8; ++i) { r[i] = lo[i]; r[i + 8] = hi[i]; }
  return r;
}

__device__ __forceinline__ v8f wmma_f16(v16h a, v16h b, v8f c) {
  return __builtin_amdgcn_wmma_f32_16x16x32_f16(false, a, false, b, (short)0, c,
                                                false, false);
}

// Registers holding one prefetched K/V tile slice (software pipeline stage).
struct Stage {
  v4f ka, kc, va, vc;
  float pen;
};

__device__ __forceinline__ void stage_load(const float* kb, const float* vb,
                                           const unsigned char* vmp, int tid,
                                           int base, Stage& s) {
  s.ka = *(const v4f*)&kb[base];      // global_load_b128 x4, in flight during
  s.kc = *(const v4f*)&kb[base + 4];  // the current tile's compute
  s.va = *(const v4f*)&vb[base];
  s.vc = *(const v4f*)&vb[base + 4];
  if (tid < 32) s.pen = vmp[tid] ? 0.f : BIG2;
}

__device__ __forceinline__ void stage_store(half_t (*sK)[72], half_t (*sVt)[40],
                                            float* sPen, int tid, int row,
                                            int col, const Stage& s) {
  v8h kh;
#pragma unroll
  for (int i = 0; i < 4; ++i) { kh[i] = (half_t)s.ka[i]; kh[i + 4] = (half_t)s.kc[i]; }
  *(v8h*)&sK[row][col] = kh;  // one ds_store_b128
#pragma unroll
  for (int i = 0; i < 4; ++i) {
    sVt[col + i][row]     = (half_t)s.va[i];  // transposed scalar stores
    sVt[col + 4 + i][row] = (half_t)s.vc[i];
  }
  if (tid < 32) sPen[tid] = s.pen;
}

// One 32-key tile: S^T = K*Q^T (lane = query column), online softmax with
// per-lane row stats, O^T += V^T * P^T.  CAUSAL selects the diagonal path.
template <bool CAUSAL>
__device__ __forceinline__ void process_tile(
    const half_t (*sK)[72], const half_t (*sVt)[40], half_t (*sPw)[40],
    const float* sPen, int n, int hi, int mOff, int jBase, int q0, float scl2,
    v16h qb0, v16h qb1, v8f& o0, v8f& o1, v8f& o2, v8f& o3, float& mrow,
    float& lrow) {
  // ---- issue ALL independent LDS operand loads up front: K rows for S^T,
  //      V^T rows for the PV GEMM.  Their latency hides under S-GEMM+softmax.
  v16h ka00 = concat8(*(const v8h*)&sK[n][mOff],           *(const v8h*)&sK[n][mOff + 16]);
  v16h ka01 = concat8(*(const v8h*)&sK[n][mOff + 32],      *(const v8h*)&sK[n][mOff + 48]);
  v16h ka10 = concat8(*(const v8h*)&sK[n + 16][mOff],      *(const v8h*)&sK[n + 16][mOff + 16]);
  v16h ka11 = concat8(*(const v8h*)&sK[n + 16][mOff + 32], *(const v8h*)&sK[n + 16][mOff + 48]);
  v16h va0 = concat8(*(const v8h*)&sVt[n][mOff],      *(const v8h*)&sVt[n][mOff + 16]);
  v16h va1 = concat8(*(const v8h*)&sVt[n + 16][mOff], *(const v8h*)&sVt[n + 16][mOff + 16]);
  v16h va2 = concat8(*(const v8h*)&sVt[n + 32][mOff], *(const v8h*)&sVt[n + 32][mOff + 16]);
  v16h va3 = concat8(*(const v8h*)&sVt[n + 48][mOff], *(const v8h*)&sVt[n + 48][mOff + 16]);

  v8f st0 = {}, st1 = {};
  st0 = wmma_f16(ka00, qb0, st0);
  st0 = wmma_f16(ka01, qb1, st0);
  st1 = wmma_f16(ka10, qb0, st1);
  st1 = wmma_f16(ka11, qb1, st1);

  // ---- per-lane penalties: this lane's 8 keys per subtile are consecutive ----
  v4f pen0a = *(const v4f*)&sPen[hi * 8];
  v4f pen0b = *(const v4f*)&sPen[hi * 8 + 4];
  v4f pen1a = *(const v4f*)&sPen[16 + hi * 8];
  v4f pen1b = *(const v4f*)&sPen[16 + hi * 8 + 4];

  float a[16];
#pragma unroll
  for (int r = 0; r < 8; ++r) {
    float p0 = (r < 4) ? pen0a[r] : pen0b[r - 4];
    float p1 = (r < 4) ? pen1a[r] : pen1b[r - 4];
    if (CAUSAL) {
      int k0 = jBase + hi * 8 + r;
      int qq = q0 + n;
      p0 += (k0 > qq) ? BIG2 : 0.f;
      p1 += (k0 + 16 > qq) ? BIG2 : 0.f;
    }
    a[r]     = __builtin_fmaf(st0[r], scl2, -p0);
    a[r + 8] = __builtin_fmaf(st1[r], scl2, -p1);
  }

  // ---- row max: balanced tree (log depth) + one cross-half shuffle ----
  float t[16];
#pragma unroll
  for (int i = 0; i < 16; ++i) t[i] = a[i];
#pragma unroll
  for (int s = 1; s < 16; s <<= 1)
#pragma unroll
    for (int i = 0; i < 16; i += 2 * s) t[i] = fmaxf(t[i], t[i + s]);
  float tm = fmaxf(t[0], __shfl_xor(t[0], 16));
  float mnew = fmaxf(mrow, tm);

  float p[16];
#pragma unroll
  for (int i = 0; i < 16; ++i) p[i] = __builtin_amdgcn_exp2f(a[i] - mnew);

  // ---- row sum: balanced tree + one cross-half shuffle ----
  float u[16];
#pragma unroll
  for (int i = 0; i < 16; ++i) u[i] = p[i];
#pragma unroll
  for (int s = 1; s < 16; s <<= 1)
#pragma unroll
    for (int i = 0; i < 16; i += 2 * s) u[i] += u[i + s];
  float rs = u[0] + __shfl_xor(u[0], 16);

  if (__any(mnew > mrow)) {  // wave-uniform: rescale only when max rose
    float alpha = __builtin_amdgcn_exp2f(mrow - mnew);
    lrow = lrow * alpha + rs;
    mrow = mnew;
#pragma unroll
    for (int r = 0; r < 8; ++r) {
      o0[r] *= alpha; o1[r] *= alpha; o2[r] *= alpha; o3[r] *= alpha;
    }
  } else {
    lrow += rs;
  }

  // ---- P^T -> LDS scratch sPw[q][k], packed f16 pairs ----
#pragma unroll
  for (int tt = 0; tt < 2; ++tt)
#pragma unroll
    for (int i = 0; i < 4; ++i) {
      f16x2 pk = __builtin_amdgcn_cvt_pkrtz(p[tt * 8 + 2 * i], p[tt * 8 + 2 * i + 1]);
      *(f16x2*)&sPw[n][tt * 16 + hi * 8 + 2 * i] = pk;
    }
  // Compiler-only fence: same-wave LDS ops execute in order in hardware, so
  // the store->load ordering is guaranteed; just stop compiler reordering.
  asm volatile("" ::: "memory");

  // ---- O^T += V^T * P^T : only pb depends on the stores above ----
  v16h pb = concat8(*(const v8h*)&sPw[n][hi * 8], *(const v8h*)&sPw[n][16 + hi * 8]);
  o0 = wmma_f16(va0, pb, o0);
  o1 = wmma_f16(va1, pb, o1);
  o2 = wmma_f16(va2, pb, o2);
  o3 = wmma_f16(va3, pb, o3);
}

__global__ __launch_bounds__(256) void fa_fwd_kernel(
    const float* __restrict__ Q, const float* __restrict__ V,
    const float* __restrict__ K, const unsigned char* __restrict__ QM,
    const unsigned char* __restrict__ VM, const float* __restrict__ SC,
    float* __restrict__ OUT) {
  __shared__ __align__(16) half_t sK[32][72];     // K tile row-major  (144B pitch)
  __shared__ __align__(16) half_t sVt[64][40];    // V tile transposed ( 80B pitch)
  __shared__ __align__(16) half_t sP[8][16][40];  // per-wave P^T scratch
  __shared__ __align__(16) float  sPen[32];       // v_mask penalty per key

  const int tid  = threadIdx.x;
  const int wave = tid >> 5;
  const int lane = tid & 31;
  const int n    = lane & 15;  // WMMA column: QUERY index in this orientation
  const int hi   = lane >> 4;
  const int mOff = hi * 8;

  const int b      = blockIdx.x >> 5;
  const int qBlock = (blockIdx.x & 31) * 128;
  const int q0     = qBlock + wave * 16;
  const float scl2 = SC[0] * LOG2E;  // log2-domain softmax

  // ---- preload Q^T in B layout: lane col = q, halves = d (table order) ----
  const float* qrow = Q + ((size_t)b * TQ + (q0 + n)) * DH;
  v16h qb0, qb1;
#pragma unroll
  for (int h = 0; h < 16; ++h) {
    qb0[h] = (half_t)qrow[(h & 8) * 2 + hi * 8 + (h & 7)];
    qb1[h] = (half_t)qrow[32 + (h & 8) * 2 + hi * 8 + (h & 7)];
  }

  v8f o0 = {}, o1 = {}, o2 = {}, o3 = {};
  float mrow = -3.0e38f, lrow = 0.f;

  const int jTilesWave  = ((q0 + 15) >> 5) + 1;
  const int jFullWave   = (q0 >= 31) ? (((q0 - 31) >> 5) + 1) : 0;  // causal-free tiles
  const int jTilesBlock = ((qBlock + 127) >> 5) + 1;

  half_t (*sPw)[40] = sP[wave];

  // ---- software pipeline: prefetch tile 0 into registers ----
  const float*         Kb  = K + (size_t)b * TV * DH;
  const float*         Vb  = V + (size_t)b * TV * DH;
  const unsigned char* VMb = VM + (size_t)b * TV;
  const int base = tid * 8;  // 2048 elems / 256 threads, 8 contiguous each
  const int row  = base >> 6;
  const int col  = base & 63;

  Stage st;
  stage_load(Kb, Vb, VMb, tid, base, st);

  for (int j32 = 0; j32 < jTilesBlock; ++j32) {
    const int jBase = j32 << 5;
    __syncthreads();  // previous tile fully consumed before restage
    stage_store(sK, sVt, sPen, tid, row, col, st);
    // prefetch next tile (clamped on last iteration); latency hides behind compute
    const int jNext = jBase + ((j32 + 1 < jTilesBlock) ? 32 : 0);
    stage_load(Kb + (size_t)jNext * DH, Vb + (size_t)jNext * DH, VMb + jNext,
               tid, base, st);
    __syncthreads();  // staged tile visible

    if (j32 < jFullWave) {          // interior: no causal compares
      process_tile<false>(sK, sVt, sPw, sPen, n, hi, mOff, jBase, q0, scl2,
                          qb0, qb1, o0, o1, o2, o3, mrow, lrow);
    } else if (j32 < jTilesWave) {  // diagonal: full masking
      process_tile<true>(sK, sVt, sPw, sPen, n, hi, mOff, jBase, q0, scl2,
                         qb0, qb1, o0, o1, o2, o3, mrow, lrow);
    }
  }

  // ---- epilogue: per-lane 1/l + q_mask, vectorized b128 stores ----
  float inv = QM[(size_t)b * TQ + q0 + n] ? (1.0f / lrow) : 0.f;
#pragma unroll
  for (int r = 0; r < 8; ++r) {
    o0[r] *= inv; o1[r] *= inv; o2[r] *= inv; o3[r] *= inv;
  }
  float* orow = OUT + ((size_t)b * TQ + q0 + n) * DH + mOff;
  *(v4f*)&orow[0]  = (v4f){o0[0], o0[1], o0[2], o0[3]};
  *(v4f*)&orow[4]  = (v4f){o0[4], o0[5], o0[6], o0[7]};
  *(v4f*)&orow[16] = (v4f){o1[0], o1[1], o1[2], o1[3]};
  *(v4f*)&orow[20] = (v4f){o1[4], o1[5], o1[6], o1[7]};
  *(v4f*)&orow[32] = (v4f){o2[0], o2[1], o2[2], o2[3]};
  *(v4f*)&orow[36] = (v4f){o2[4], o2[5], o2[6], o2[7]};
  *(v4f*)&orow[48] = (v4f){o3[0], o3[1], o3[2], o3[3]};
  *(v4f*)&orow[52] = (v4f){o3[4], o3[5], o3[6], o3[7]};
}

extern "C" void kernel_launch(void* const* d_in, const int* in_sizes, int n_in,
                              void* d_out, int out_size, void* d_ws, size_t ws_size,
                              hipStream_t stream) {
  (void)in_sizes; (void)n_in; (void)out_size; (void)d_ws; (void)ws_size;
  const float*         q  = (const float*)d_in[0];
  const float*         v  = (const float*)d_in[1];
  const float*         k  = (const float*)d_in[2];
  const unsigned char* qm = (const unsigned char*)d_in[3];
  const unsigned char* vm = (const unsigned char*)d_in[4];
  const float*         sc = (const float*)d_in[5];
  float*               out = (float*)d_out;

  dim3 grid(4 * (TQ / 128));  // B * Tq/128 = 128 blocks
  dim3 block(256);            // 8 wave32 waves, 16 Q rows each
  fa_fwd_kernel<<<grid, block, 0, stream>>>(q, v, k, qm, vm, sc, out);
}
